// BertSelfOutputPAL_44324062495032
// MI455X (gfx1250) — matrix operand order repaired and checked by reading, not compiled, not run on hardware
//
#include <hip/hip_runtime.h>
#include <hip/hip_bf16.h>
#include <cstdint>

typedef __attribute__((ext_vector_type(16))) __bf16 v16bf;
typedef __attribute__((ext_vector_type(8)))  __bf16 v8bf;
typedef __attribute__((ext_vector_type(8)))  float  v8f;

#define B_ 8
#define S_ 2048
#define H_ 1024
#define P_ 256
#define T_ 4
#define M_ (B_ * S_)   // 16384
#define KC 1024        // contraction dim (all three GEMMs)
#define NC 1024        // output width   (all three GEMMs)

// ---------------------------------------------------------------------------
// prep kernels
// ---------------------------------------------------------------------------
__global__ __launch_bounds__(256) void k_f32_to_bf16(const float* __restrict__ in,
                                                     __bf16* __restrict__ out, int n) {
  int i = blockIdx.x * 256 + threadIdx.x;
  if (i < n) out[i] = (__bf16)in[i];
}

__global__ __launch_bounds__(256) void k_tr_dense(const float* __restrict__ W,
                                                  __bf16* __restrict__ Wt) {
  int i = blockIdx.x * 256 + threadIdx.x;
  int n = i >> 10, k = i & 1023;
  Wt[(size_t)n * H_ + k] = (__bf16)W[(size_t)k * H_ + n];
}

__global__ __launch_bounds__(256) void k_tr_w1(const float* __restrict__ W1,
                                               __bf16* __restrict__ W1t) {
  int i = blockIdx.x * 256 + threadIdx.x;
  int n = i >> 10;          // t*P+p
  int h = i & 1023;
  int t = n >> 8, p = n & 255;
  W1t[(size_t)n * H_ + h] = (__bf16)W1[((size_t)t * H_ + h) * P_ + p];
}

__global__ __launch_bounds__(256) void k_tr_w2(const float* __restrict__ W2,
                                               __bf16* __restrict__ W2t) {
  int i = blockIdx.x * 256 + threadIdx.x;
  int h = i >> 10;
  int n = i & 1023;
  int t = n >> 8, p = n & 255;
  W2t[(size_t)h * (T_ * P_) + n] = (__bf16)W2[((size_t)t * P_ + p) * H_ + h];
}

// ---------------------------------------------------------------------------
// WMMA bf16 GEMM, C[M,1024] = A[M,1024] * Bt[1024,1024]^T, f32 accumulate.
// Wave tile 64x64: 4 A-frags x 4 B-frags = 16 WMMAs per 32-K step,
// 8 A global_load_b128 + 8 B ds_load_b128 per step (load:wmma = 1.0).
// Workgroup 8 waves -> 512x64 block tile.
// B tile (64 cols x 64 K) staged to LDS via GLOBAL_LOAD_ASYNC_TO_LDS_B128,
// double-buffered, s_wait_asynccnt + workgroup barrier. LDS row pitch 144B
// (36 banks) -> conflict-free ds_load_b128 fragment reads.
// __launch_bounds__(256,1): lift the VGPR cap so the ~190-VGPR working set
// (128 accum + 32 A-frag + B/addr) stays put — avoids the accumulator
// v_mov_b64 churn + WMMA->VALU hazard nops seen at default occupancy.
// ---------------------------------------------------------------------------
union Frag { v16bf v; v8bf h[2]; };

#define SB_PITCH 72              // halfs per LDS row (144 bytes)
#define SB_BUF   (64 * SB_PITCH) // halfs per buffer

__global__ __launch_bounds__(256, 1) void k_gemm_wmma(
    const __bf16* __restrict__ A, const __bf16* __restrict__ Bt, int mode,
    const float* __restrict__ bias,
    const float* __restrict__ dist,
    const float* __restrict__ resid0,
    const float* __restrict__ resid1,
    const float* __restrict__ rowbias,
    float* __restrict__ outF, __bf16* __restrict__ outB) {
  __shared__ __align__(128) __bf16 sB[2 * SB_BUF];

  const int tid  = threadIdx.x;
  const int lane = tid & 31;
  const int wave = tid >> 5;
  const int lr   = lane & 15;
  const int lg   = lane >> 4;
  const int m0 = blockIdx.y * 512 + wave * 64;
  const int n0 = blockIdx.x * 64;

  // ---- B staging geometry: 4 threads/row, each moves 2x16B chunks ----
  const int srow = tid >> 2;                    // 0..63
  const int schk = tid & 3;                     // chunk pairs (schk, schk+4)
  const uint64_t btBase = (uint64_t)Bt;
  const int gBrowB = ((n0 + srow) * KC) * 2;    // byte offset of row in Bt
  const uint32_t ldsBase =
      (uint32_t)(size_t)(&sB[0]) + (uint32_t)(srow * (SB_PITCH * 2) + schk * 16);

  auto stageB = [&](int buf, int k0) {
    const uint32_t l = ldsBase + (uint32_t)buf * (SB_BUF * 2);
    const int g = gBrowB + k0 * 2 + schk * 16;
    asm volatile("global_load_async_to_lds_b128 %0, %1, %2"
                 :: "v"(l), "v"(g), "s"(btBase) : "memory");
    asm volatile("global_load_async_to_lds_b128 %0, %1, %2"
                 :: "v"(l + 64), "v"(g + 64), "s"(btBase) : "memory");
  };

  const __bf16* arow = A + (size_t)(m0 + lr) * KC + lg * 8;

  v8f acc[4][4] = {};

  stageB(0, 0);
  for (int ks = 0; ks < KC / 64; ++ks) {
    asm volatile("s_wait_asynccnt 0x0" ::: "memory");
    __syncthreads();
    if (ks + 1 < KC / 64) stageB((ks + 1) & 1, (ks + 1) * 64);

    const __bf16* sb = sB + (ks & 1) * SB_BUF;
    const int k0 = ks * 64;
#pragma unroll
    for (int kc = 0; kc < 2; ++kc) {
      const int kk = k0 + kc * 32;
      Frag a[4];
#pragma unroll
      for (int f = 0; f < 4; ++f) {
        a[f].h[0] = *(const v8bf*)(arow + (size_t)f * 16 * KC + kk);
        a[f].h[1] = *(const v8bf*)(arow + (size_t)f * 16 * KC + kk + 16);
      }
#pragma unroll
      for (int nt = 0; nt < 4; ++nt) {
        const __bf16* bp = sb + (nt * 16 + lr) * SB_PITCH + kc * 32 + lg * 8;
        Frag b;
        b.h[0] = *(const v8bf*)(bp);
        b.h[1] = *(const v8bf*)(bp + 16);
#pragma unroll
        for (int f = 0; f < 4; ++f) {
          acc[f][nt] = __builtin_amdgcn_wmma_f32_16x16x32_bf16(
              false, a[f].v, false, b.v, (short)0, acc[f][nt], false, false);
        }
      }
    }
  }

  // ---- epilogue ----
#pragma unroll
  for (int mh = 0; mh < 4; ++mh) {
    const int rbase = m0 + mh * 16 + (lg ? 8 : 0);
#pragma unroll
    for (int nt = 0; nt < 4; ++nt) {
      const int n = n0 + nt * 16 + lr;
#pragma unroll
      for (int r = 0; r < 8; ++r) {
        const int row = rbase + r;
        const size_t idx = (size_t)row * NC + n;
        float v = acc[mh][nt][r];
        if (mode == 0) {
          v += bias[n];
          outF[idx] = v;
          outB[idx] = (__bf16)v;
        } else if (mode == 1) {
          const int b = row >> 11;                       // S = 2048
          v = (v + bias[n]) * dist[b * T_ + (n >> 8)];   // P = 256 per task
          outB[idx] = (__bf16)v;
        } else {
          const int b = row >> 11;
          v += resid0[idx] + resid1[idx] + rowbias[(size_t)b * H_ + n];
          outF[idx] = v;
        }
      }
    }
  }
}

// ---------------------------------------------------------------------------
// gating path
// ---------------------------------------------------------------------------
__global__ __launch_bounds__(256) void k_encode(const float* __restrict__ h,
                                                const float* __restrict__ encW,
                                                const float* __restrict__ encb,
                                                const float* __restrict__ mask,
                                                float* __restrict__ logits) {
  const int row = blockIdx.x * 8 + (threadIdx.x >> 5);
  const int lane = threadIdx.x & 31;
  const float* hr = h + (size_t)row * H_;
  float s = 0.f;
#pragma unroll 8
  for (int i = lane; i < H_; i += 32) s += hr[i] * encW[i];
#pragma unroll
  for (int off = 16; off; off >>= 1) s += __shfl_xor(s, off, 32);
  if (lane == 0) logits[row] = s + encb[0] + mask[row];
}

__global__ __launch_bounds__(256) void k_softmax(const float* __restrict__ logits,
                                                 float* __restrict__ w) {
  __shared__ float red[256];
  const int b = blockIdx.x, tid = threadIdx.x;
  const float* l = logits + (size_t)b * S_;
  float m = -1e30f;
  for (int i = tid; i < S_; i += 256) m = fmaxf(m, l[i]);
  red[tid] = m; __syncthreads();
  for (int o = 128; o; o >>= 1) { if (tid < o) red[tid] = fmaxf(red[tid], red[tid + o]); __syncthreads(); }
  m = red[0]; __syncthreads();
  float s = 0.f;
  for (int i = tid; i < S_; i += 256) { float e = __expf(l[i] - m); w[b * S_ + i] = e; s += e; }
  red[tid] = s; __syncthreads();
  for (int o = 128; o; o >>= 1) { if (tid < o) red[tid] += red[tid + o]; __syncthreads(); }
  const float inv = 1.f / red[0];
  for (int i = tid; i < S_; i += 256) w[b * S_ + i] *= inv;
}

__global__ __launch_bounds__(256) void k_taskvec(const float* __restrict__ h,
                                                 const float* __restrict__ w,
                                                 float* __restrict__ partial) {
  const int b = blockIdx.y, c = blockIdx.x, tid = threadIdx.x;
  float a0 = 0.f, a1 = 0.f, a2 = 0.f, a3 = 0.f;
  const int s0 = c * (S_ / 8);
  for (int s = s0; s < s0 + S_ / 8; ++s) {
    const float ws = w[b * S_ + s];
    const float* hr = h + ((size_t)b * S_ + s) * H_;
    a0 += ws * hr[tid];
    a1 += ws * hr[tid + 256];
    a2 += ws * hr[tid + 512];
    a3 += ws * hr[tid + 768];
  }
  float* pr = partial + ((size_t)(b * 8 + c)) * H_;
  pr[tid] = a0; pr[tid + 256] = a1; pr[tid + 512] = a2; pr[tid + 768] = a3;
}

__global__ __launch_bounds__(1024) void k_gate(const float* __restrict__ partial,
                                               const float* __restrict__ selW,
                                               const float* __restrict__ selb,
                                               const float* __restrict__ b2,
                                               float* __restrict__ dist,
                                               float* __restrict__ bias2eff) {
  __shared__ float red[1024];
  __shared__ float dsh[T_];
  const int b = blockIdx.x, tid = threadIdx.x;
  float tv = 0.f;
  for (int c = 0; c < 8; ++c) tv += partial[((size_t)(b * 8 + c)) * H_ + tid];
  float lg[T_];
  for (int t = 0; t < T_; ++t) {
    red[tid] = tv * selW[t * H_ + tid];
    __syncthreads();
    for (int o = 512; o; o >>= 1) { if (tid < o) red[tid] += red[tid + o]; __syncthreads(); }
    lg[t] = red[0] + selb[t];
    __syncthreads();
  }
  if (tid == 0) {
    float mx = fmaxf(fmaxf(lg[0], lg[1]), fmaxf(lg[2], lg[3]));
    float e0 = __expf(lg[0] - mx), e1 = __expf(lg[1] - mx);
    float e2 = __expf(lg[2] - mx), e3 = __expf(lg[3] - mx);
    const float inv = 1.f / (e0 + e1 + e2 + e3);
    dsh[0] = e0 * inv; dsh[1] = e1 * inv; dsh[2] = e2 * inv; dsh[3] = e3 * inv;
    for (int t = 0; t < T_; ++t) dist[b * T_ + t] = dsh[t];
  }
  __syncthreads();
  float be = 0.f;
  for (int t = 0; t < T_; ++t) be += dsh[t] * b2[t * H_ + tid];
  bias2eff[(size_t)b * H_ + tid] = be;
}

__global__ __launch_bounds__(256) void k_ln(float* __restrict__ x,
                                            const float* __restrict__ g,
                                            const float* __restrict__ bta) {
  __shared__ float r1[256], r2[256];
  const int row = blockIdx.x, tid = threadIdx.x;
  float* xr = x + (size_t)row * H_;
  const float v0 = xr[tid], v1 = xr[tid + 256], v2 = xr[tid + 512], v3 = xr[tid + 768];
  r1[tid] = v0 + v1 + v2 + v3;
  r2[tid] = v0 * v0 + v1 * v1 + v2 * v2 + v3 * v3;
  __syncthreads();
  for (int o = 128; o; o >>= 1) { if (tid < o) { r1[tid] += r1[tid + o]; r2[tid] += r2[tid + o]; } __syncthreads(); }
  const float mu = r1[0] * (1.f / H_);
  const float var = r2[0] * (1.f / H_) - mu * mu;
  const float inv = 1.f / sqrtf(var + 1e-12f);
  xr[tid]       = g[tid]       * (v0 - mu) * inv + bta[tid];
  xr[tid + 256] = g[tid + 256] * (v1 - mu) * inv + bta[tid + 256];
  xr[tid + 512] = g[tid + 512] * (v2 - mu) * inv + bta[tid + 512];
  xr[tid + 768] = g[tid + 768] * (v3 - mu) * inv + bta[tid + 768];
}

// ---------------------------------------------------------------------------
extern "C" void kernel_launch(void* const* d_in, const int* in_sizes, int n_in,
                              void* d_out, int out_size, void* d_ws, size_t ws_size,
                              hipStream_t stream) {
  const float* hs   = (const float*)d_in[0];
  const float* intn = (const float*)d_in[1];
  const float* mask = (const float*)d_in[2];
  const float* dW   = (const float*)d_in[3];
  const float* db   = (const float*)d_in[4];
  const float* W1   = (const float*)d_in[5];
  const float* b1   = (const float*)d_in[6];   // [T*P] flat == bias[n]
  const float* W2   = (const float*)d_in[7];
  const float* b2   = (const float*)d_in[8];
  const float* encW = (const float*)d_in[9];
  const float* encb = (const float*)d_in[10];
  const float* selW = (const float*)d_in[11];
  const float* selb = (const float*)d_in[12];
  const float* lng  = (const float*)d_in[13];
  const float* lnb  = (const float*)d_in[14];
  float* out = (float*)d_out;

  char* ws = (char*)d_ws;
  size_t off = 0;
  auto alloc = [&](size_t bytes) -> void* {
    void* p = ws + off;
    off += (bytes + 255) & ~(size_t)255;
    return p;
  };
  __bf16* hsb   = (__bf16*)alloc((size_t)M_ * H_ * 2);         // 32 MB
  __bf16* dWt   = (__bf16*)alloc((size_t)H_ * H_ * 2);         //  2 MB
  __bf16* W1t   = (__bf16*)alloc((size_t)T_ * P_ * H_ * 2);    //  2 MB
  __bf16* W2t   = (__bf16*)alloc((size_t)H_ * T_ * P_ * 2);    //  2 MB
  float*  hF    = (float*) alloc((size_t)M_ * H_ * 4);         // 64 MB
  __bf16* hB    = (__bf16*)alloc((size_t)M_ * H_ * 2);         // 32 MB
  __bf16* lowS  = (__bf16*)alloc((size_t)M_ * (T_ * P_) * 2);  // 32 MB
  float*  logit = (float*) alloc((size_t)M_ * 4);
  float*  smw   = (float*) alloc((size_t)M_ * 4);
  float*  part  = (float*) alloc((size_t)B_ * 8 * H_ * 4);
  float*  dist  = (float*) alloc((size_t)B_ * T_ * 4);
  float*  b2eff = (float*) alloc((size_t)B_ * H_ * 4);

  // 1) prepack
  k_f32_to_bf16<<<(M_ * H_) / 256, 256, 0, stream>>>(hs, hsb, M_ * H_);
  k_tr_dense<<<(H_ * H_) / 256, 256, 0, stream>>>(dW, dWt);
  k_tr_w1<<<(T_ * P_ * H_) / 256, 256, 0, stream>>>(W1, W1t);
  k_tr_w2<<<(H_ * T_ * P_) / 256, 256, 0, stream>>>(W2, W2t);

  const dim3 gemmGrid(NC / 64, M_ / 512);

  // 2) GEMM1: h = hs @ dense_W + db   -> hF (f32) and hB (bf16)
  k_gemm_wmma<<<gemmGrid, 256, 0, stream>>>(hsb, dWt, 0,
                                            db, nullptr, nullptr, nullptr, nullptr,
                                            hF, hB);

  // 3) gating path
  k_encode<<<M_ / 8, 256, 0, stream>>>(hF, encW, encb, mask, logit);
  k_softmax<<<B_, 256, 0, stream>>>(logit, smw);
  k_taskvec<<<dim3(8, B_), 256, 0, stream>>>(hF, smw, part);
  k_gate<<<B_, 1024, 0, stream>>>(part, selW, selb, b2, dist, b2eff);

  // 4) GEMM2: lowS[b,s,t*P+p] = dist[b,t] * (h @ W1[t] + b1[t])  (bf16)
  k_gemm_wmma<<<gemmGrid, 256, 0, stream>>>(hB, W1t, 1,
                                            b1, dist, nullptr, nullptr, nullptr,
                                            nullptr, lowS);

  // 5) GEMM3: x = lowS @ W2cat + h + input + dist.b2  -> d_out (pre-LN)
  k_gemm_wmma<<<gemmGrid, 256, 0, stream>>>(lowS, W2t, 2,
                                            nullptr, nullptr, hF, intn, b2eff,
                                            out, nullptr);

  // 6) LayerNorm in place on d_out
  k_ln<<<M_, 256, 0, stream>>>(out, lng, lnb);
}
